// CascadeRoIHeadWarper_2765958939466
// MI455X (gfx1250) — matrix operand order repaired and checked by reading, not compiled, not run on hardware
//
#include <hip/hip_runtime.h>

// ---------------------------------------------------------------------------
// Cascade R-CNN RoI head for MI455X (gfx1250, wave32, WMMA bf16 + TDM)
// ---------------------------------------------------------------------------

#define NUM_ROIS   512
#define NUM_CH     256
#define POOL       7
#define FEAT_D     (NUM_CH * POOL * POOL)   // 12544
#define FC_DIM     1024
#define NUM_CLS    80
#define NUM_CLS1   81
#define MAX_DET    100
#define MAX_RATIO_F 4.135166556742356f      // |log(16/1000)|

#define KSTEP      64
#define TPITCH     72                       // 64 bf16 + 8 pad (144 B, 16-aligned)

typedef __bf16 bf16_t;
typedef bf16_t v16bf __attribute__((ext_vector_type(16)));
typedef float  v8f   __attribute__((ext_vector_type(8)));
typedef unsigned int u32x4v __attribute__((ext_vector_type(4)));
typedef int          i32x4v __attribute__((ext_vector_type(4)));
typedef int          i32x8v __attribute__((ext_vector_type(8)));

#if __has_builtin(__builtin_amdgcn_tensor_load_to_lds) && \
    __has_builtin(__builtin_amdgcn_s_wait_tensorcnt)
#define USE_TDM 1
#else
#define USE_TDM 0
#endif

union FragBF { v16bf v; unsigned int u[8]; };

__device__ __forceinline__ unsigned short f2bf(float f) {
    unsigned int u = __float_as_uint(f);
    unsigned int r = u + 0x7FFFu + ((u >> 16) & 1u);   // round-to-nearest-even
    return (unsigned short)(r >> 16);
}
__device__ __forceinline__ float bf2f(unsigned short h) {
    return __uint_as_float(((unsigned int)h) << 16);
}

__device__ __forceinline__ void delta2bbox_one(const float* r, const float* d,
                                               float s0, float s1, float s2, float s3,
                                               float* o) {
    float dx = d[0] * s0, dy = d[1] * s1;
    float dw = fminf(fmaxf(d[2] * s2, -MAX_RATIO_F), MAX_RATIO_F);
    float dh = fminf(fmaxf(d[3] * s3, -MAX_RATIO_F), MAX_RATIO_F);
    float px = (r[0] + r[2]) * 0.5f, py = (r[1] + r[3]) * 0.5f;
    float pw = r[2] - r[0], ph = r[3] - r[1];
    float gw = pw * __expf(dw), gh = ph * __expf(dh);
    float gx = px + pw * dx, gy = py + ph * dy;
    o[0] = gx - gw * 0.5f; o[1] = gy - gh * 0.5f;
    o[2] = gx + gw * 0.5f; o[3] = gy + gh * 0.5f;
}

// ---------------------------------------------------------------------------
// RoI align (level-mapped, 7x7, 2x2 samples) -> bf16 activation matrix X
// ---------------------------------------------------------------------------
__global__ __launch_bounds__(256)
void roi_align_kernel(const float* __restrict__ f0, const float* __restrict__ f1,
                      const float* __restrict__ f2, const float* __restrict__ f3,
                      const float* __restrict__ rois, unsigned short* __restrict__ X) {
    int roi = blockIdx.x;
    float x1 = rois[roi * 4 + 0], y1 = rois[roi * 4 + 1];
    float x2 = rois[roi * 4 + 2], y2 = rois[roi * 4 + 3];
    float area = fmaxf((x2 - x1) * (y2 - y1), 0.0f);
    float sdim = sqrtf(area);
    float lv = floorf(__log2f(sdim / 56.0f + 1e-6f));
    int lvl = (int)fminf(fmaxf(lv, 0.0f), 3.0f);

    const float* feat; int H, W; float stride;
    if      (lvl == 0) { feat = f0; H = 200; W = 336; stride = 4.0f;  }
    else if (lvl == 1) { feat = f1; H = 100; W = 168; stride = 8.0f;  }
    else if (lvl == 2) { feat = f2; H = 50;  W = 84;  stride = 16.0f; }
    else               { feat = f3; H = 25;  W = 42;  stride = 32.0f; }

    float scale = 1.0f / stride;
    float rx1 = x1 * scale - 0.5f, ry1 = y1 * scale - 0.5f;
    float rx2 = x2 * scale - 0.5f, ry2 = y2 * scale - 0.5f;
    float bw = (rx2 - rx1) * (1.0f / POOL);
    float bh = (ry2 - ry1) * (1.0f / POOL);
    float Hf = (float)H, Wf = (float)W;

    for (int o = threadIdx.x; o < FEAT_D; o += 256) {
        int c = o / 49, p = o % 49, ph = p / 7, pw = p % 7;
        const float* fc = feat + (size_t)c * H * W;
        float acc = 0.0f;
        #pragma unroll
        for (int iy = 0; iy < 2; iy++) {
            float yy = ry1 + ((float)ph + ((float)iy + 0.5f) * 0.5f) * bh;
            #pragma unroll
            for (int ix = 0; ix < 2; ix++) {
                float xx = rx1 + ((float)pw + ((float)ix + 0.5f) * 0.5f) * bw;
                float valid = (yy > -1.0f && yy < Hf && xx > -1.0f && xx < Wf) ? 1.0f : 0.0f;
                float yc = fminf(fmaxf(yy, 0.0f), Hf - 1.0f);
                float xc = fminf(fmaxf(xx, 0.0f), Wf - 1.0f);
                float y0f = floorf(yc), x0f = floorf(xc);
                int y0 = (int)y0f, x0 = (int)x0f;
                int y1i = min(y0 + 1, H - 1), x1i = min(x0 + 1, W - 1);
                float ly = yc - y0f, lx = xc - x0f;
                float v00 = fc[y0 * W + x0],  v01 = fc[y0 * W + x1i];
                float v10 = fc[y1i * W + x0], v11 = fc[y1i * W + x1i];
                float val = (v00 * (1.0f - ly) * (1.0f - lx) + v01 * (1.0f - ly) * lx +
                             v10 * ly * (1.0f - lx) + v11 * ly * lx) * valid;
                acc += val;
            }
        }
        X[(size_t)roi * FEAT_D + o] = f2bf(acc * 0.25f);
    }
}

// ---------------------------------------------------------------------------
// WMMA bf16 GEMM:  Y[N,M] = relu(X[N,K] * W[K,M] + bias)
// X tile: TDM (tensor_load_to_lds) into padded LDS, double buffered.
// W tile: fp32 global float4 loads -> bf16 pack -> LDS, double buffered.
// Block: 128 thr = 4 waves, 64x64 tile; 8x v_wmma_f32_16x16x32_bf16 / K-step.
// ---------------------------------------------------------------------------
__global__ __launch_bounds__(128)
void gemm_bf16_kernel(const unsigned short* __restrict__ X,
                      const float* __restrict__ W,
                      const float* __restrict__ bias,
                      unsigned short* __restrict__ Y,
                      int N, int K, int M, int relu) {
    __shared__ __align__(16) unsigned short Xt[2][64][TPITCH];
    __shared__ __align__(16) unsigned short Wt[2][64][TPITCH];

    const int tid  = threadIdx.x;
    const int lane = tid & 31;
    const int wave = tid >> 5;
    const int wrow = wave >> 1;           // 0..1
    const int wcol = wave & 1;            // 0..1
    const int rowBase = blockIdx.y * 64;
    const int colBase = blockIdx.x * 64;
    const int T = K / KSTEP;

    const int n0  = (tid & 15) * 4;       // W: 4 consecutive cols
    const int pi0 = tid >> 4;             // W: k-pair group 0..7

    v8f acc[2][2] = {};
    float4 wreg[8];

    auto loadW = [&](int kt) {
        const float* Wg = W + (size_t)(kt * KSTEP) * M + colBase + n0;
        #pragma unroll
        for (int j = 0; j < 4; j++) {
            int kk = (pi0 + j * 8) * 2;
            wreg[2 * j]     = *(const float4*)(Wg + (size_t)kk * M);
            wreg[2 * j + 1] = *(const float4*)(Wg + (size_t)(kk + 1) * M);
        }
    };
    auto storeW = [&](int nb) {
        #pragma unroll
        for (int j = 0; j < 4; j++) {
            int kk = (pi0 + j * 8) * 2;
            float4 a = wreg[2 * j], b = wreg[2 * j + 1];
            const float* ap = (const float*)&a;
            const float* bp = (const float*)&b;
            #pragma unroll
            for (int i = 0; i < 4; i++) {
                unsigned int pk = (unsigned int)f2bf(ap[i]) |
                                  ((unsigned int)f2bf(bp[i]) << 16);
                *(unsigned int*)&Wt[nb][n0 + i][kk] = pk;
            }
        }
    };

#if USE_TDM
    // 2D TDM descriptor: 64 rows x 128 B, global row stride = K elements,
    // LDS destination padded by 16 B per 128-B row (pitch 144 B).
    auto tdmLoadX = [&](int kt, int nb) {
        unsigned long long ga =
            (unsigned long long)(X + (size_t)rowBase * K + (size_t)kt * KSTEP);
        unsigned lds = (unsigned)(size_t)&Xt[nb][0][0];
        u32x4v g0;
        g0[0] = 1u;                                   // count = 1 valid descriptor
        g0[1] = lds;                                  // lds_addr
        g0[2] = (unsigned)(ga & 0xffffffffu);         // global_addr[31:0]
        g0[3] = (unsigned)((ga >> 32) & 0x01ffffffu)  // global_addr[56:32]
              | 0x80000000u;                          // type = 2 (image)
        i32x8v g1;
        g1[0] = (1 << 16)                             // data_size = 2 B
              | (1 << 20)                             // pad_enable
              | (4 << 22)                             // pad_interval = 32 DWORDs
              | (3 << 25);                            // pad_amount  = 4 DWORDs
        g1[1] = (int)(64u << 16);                     // tensor_dim0[15:0]
        g1[2] = (int)(64u << 16);                     // tensor_dim0[31:16]=0 | tensor_dim1[15:0]
        g1[3] = (int)(64u << 16);                     // tensor_dim1[31:16]=0 | tile_dim0
        g1[4] = 64;                                   // tile_dim1 | tile_dim2=0
        g1[5] = (int)(unsigned)K;                     // tensor_dim0_stride[31:0]
        g1[6] = 0;
        g1[7] = 0;
        i32x4v z4 = {0, 0, 0, 0};
#if __clang_major__ >= 23
        i32x8v z8 = {0, 0, 0, 0, 0, 0, 0, 0};
        __builtin_amdgcn_tensor_load_to_lds(g0, g1, z4, z4, z8, 0);
#else
        __builtin_amdgcn_tensor_load_to_lds(g0, g1, z4, z4, 0);
#endif
    };
#else
    uint4 xreg[4];
    auto loadX = [&](int kt) {
        const uint4* g = (const uint4*)(X + (size_t)(rowBase + (tid >> 1)) * K +
                                        (size_t)kt * KSTEP) + (tid & 1) * 4;
        xreg[0] = g[0]; xreg[1] = g[1]; xreg[2] = g[2]; xreg[3] = g[3];
    };
    auto storeX = [&](int nb) {
        uint4* d = (uint4*)&Xt[nb][tid >> 1][(tid & 1) * 32];
        d[0] = xreg[0]; d[1] = xreg[1]; d[2] = xreg[2]; d[3] = xreg[3];
    };
#endif

    // ---- prologue: tile 0 ----
#if USE_TDM
    if (wave == 0) tdmLoadX(0, 0);
#else
    loadX(0); storeX(0);
#endif
    loadW(0);
    storeW(0);

    // ---- main loop: 1 barrier / K-step, double buffered ----
    for (int kt = 0; kt < T; kt++) {
        const int buf = kt & 1;
        if (kt + 1 < T) {
            loadW(kt + 1);                    // global fp32 loads in flight
#if !USE_TDM
            loadX(kt + 1);
#endif
            if (kt + 2 < T)                   // GL2 prefetch two tiles ahead
                __builtin_prefetch(W + (size_t)((kt + 2) * KSTEP) * M + colBase + n0, 0, 1);
        }
#if USE_TDM
        if (wave == 0) __builtin_amdgcn_s_wait_tensorcnt(0);   // tile kt landed
#endif
        __syncthreads();                      // tile kt visible; all done with buf^1
#if USE_TDM
        if (kt + 1 < T && wave == 0) tdmLoadX(kt + 1, buf ^ 1);
#endif

        const int hi = lane >> 4;
        #pragma unroll
        for (int ks = 0; ks < 2; ks++) {
            FragBF a[2], b[2];
            #pragma unroll
            for (int ti = 0; ti < 2; ti++) {
                int row = wrow * 32 + ti * 16 + (lane & 15);
                #pragma unroll
                for (int i = 0; i < 8; i++) {
                    int kb = ks * 32 + ((i >= 4) ? 16 : 0) + (i & 3) * 2 + hi * 8;
                    a[ti].u[i] = *(const unsigned int*)&Xt[buf][row][kb];
                }
            }
            #pragma unroll
            for (int tj = 0; tj < 2; tj++) {
                int col = wcol * 32 + tj * 16 + (lane & 15);
                #pragma unroll
                for (int i = 0; i < 8; i++)
                    b[tj].u[i] = *(const unsigned int*)&Wt[buf][col][ks * 32 + ((i >= 4) ? 16 : 0) + (i & 3) * 2 + hi * 8];
            }
            #pragma unroll
            for (int ti = 0; ti < 2; ti++)
                #pragma unroll
                for (int tj = 0; tj < 2; tj++)
                    acc[ti][tj] = __builtin_amdgcn_wmma_f32_16x16x32_bf16(
                        false, a[ti].v, false, b[tj].v,
                        (short)0, acc[ti][tj], false, false);
        }

        if (kt + 1 < T) {                     // fill buf^1 while others compute
#if !USE_TDM
            storeX(buf ^ 1);
#endif
            storeW(buf ^ 1);
        }
    }

    // ---- epilogue: bias + relu, per C/D VGPR layout ----
    const int n  = lane & 15;
    const int mo = (lane >> 4) * 8;
    #pragma unroll
    for (int ti = 0; ti < 2; ti++) {
        #pragma unroll
        for (int tj = 0; tj < 2; tj++) {
            int col = colBase + wcol * 32 + tj * 16 + n;
            float bv = bias[col];
            #pragma unroll
            for (int r = 0; r < 8; r++) {
                int row = rowBase + wrow * 32 + ti * 16 + r + mo;
                float v = acc[ti][tj][r] + bv;
                if (relu) v = fmaxf(v, 0.0f);
                Y[(size_t)row * M + col] = f2bf(v);
            }
        }
    }
}

// ---------------------------------------------------------------------------
// cls / reg heads: one block per RoI. cls logits accumulate across stages.
// ---------------------------------------------------------------------------
__global__ __launch_bounds__(256)
void heads_kernel(const unsigned short* __restrict__ H2,
                  const float* __restrict__ cls_w, const float* __restrict__ cls_b,
                  const float* __restrict__ reg_w, const float* __restrict__ reg_b,
                  float* __restrict__ clsSum, float* __restrict__ bboxPred, int stage) {
    __shared__ float h[FC_DIM];
    int roi = blockIdx.x, tid = threadIdx.x;
    for (int i = tid; i < FC_DIM; i += 256)
        h[i] = bf2f(H2[(size_t)roi * FC_DIM + i]);
    __syncthreads();
    if (tid < NUM_CLS1) {
        float s = 0.0f;
        for (int k = 0; k < FC_DIM; k++)
            s += h[k] * cls_w[(size_t)k * NUM_CLS1 + tid];
        s += cls_b[tid];
        if (stage == 0) clsSum[roi * NUM_CLS1 + tid] = s;
        else            clsSum[roi * NUM_CLS1 + tid] += s;
    } else if (tid < NUM_CLS1 + 4) {
        int j = tid - NUM_CLS1;
        float s = 0.0f;
        for (int k = 0; k < FC_DIM; k++)
            s += h[k] * reg_w[(size_t)k * 4 + j];
        bboxPred[roi * 4 + j] = s + reg_b[j];
    }
}

// ---------------------------------------------------------------------------
// RoI refinement between stages
// ---------------------------------------------------------------------------
__global__ __launch_bounds__(256)
void delta2bbox_kernel(const float* __restrict__ roisIn, const float* __restrict__ deltas,
                       float* __restrict__ roisOut,
                       float s0, float s1, float s2, float s3) {
    int roi = blockIdx.x * 256 + threadIdx.x;
    if (roi >= NUM_ROIS) return;
    delta2bbox_one(&roisIn[roi * 4], &deltas[roi * 4], s0, s1, s2, s3, &roisOut[roi * 4]);
}

// ---------------------------------------------------------------------------
// Average logits -> softmax -> per-class scores (transposed) + final boxes
// ---------------------------------------------------------------------------
__global__ __launch_bounds__(128)
void softmax_final_kernel(const float* __restrict__ clsSum,
                          const float* __restrict__ rois,
                          const float* __restrict__ bboxPred,
                          float* __restrict__ scoresT,   // [80][512]
                          float* __restrict__ bboxes) {  // [512][4]
    int roi = blockIdx.x * 128 + threadIdx.x;
    if (roi >= NUM_ROIS) return;
    float l[NUM_CLS1];
    float m = -1e30f;
    for (int c = 0; c < NUM_CLS1; c++) {
        l[c] = clsSum[roi * NUM_CLS1 + c] * (1.0f / 3.0f);
        m = fmaxf(m, l[c]);
    }
    float sum = 0.0f;
    for (int c = 0; c < NUM_CLS1; c++) { l[c] = __expf(l[c] - m); sum += l[c]; }
    float inv = 1.0f / sum;
    for (int c = 0; c < NUM_CLS; c++)
        scoresT[c * NUM_ROIS + roi] = l[c] * inv;
    delta2bbox_one(&rois[roi * 4], &bboxPred[roi * 4],
                   0.033f, 0.033f, 0.067f, 0.067f, &bboxes[roi * 4]);
}

// ---------------------------------------------------------------------------
// Per-class NMS: 100 sequential argmax+suppress rounds (matches lax.scan)
// ---------------------------------------------------------------------------
__global__ __launch_bounds__(256)
void nms_kernel(const float* __restrict__ bboxes, const float* __restrict__ scoresT,
                int* __restrict__ selI, float* __restrict__ selS) {
    __shared__ float bx[NUM_ROIS][4];
    __shared__ float ar[NUM_ROIS];
    __shared__ float sc[NUM_ROIS];
    __shared__ float rv[256];
    __shared__ int   ri[256];
    int cls = blockIdx.x, tid = threadIdx.x;
    for (int i = tid; i < NUM_ROIS; i += 256) {
        float b0 = bboxes[i * 4 + 0], b1 = bboxes[i * 4 + 1];
        float b2 = bboxes[i * 4 + 2], b3 = bboxes[i * 4 + 3];
        bx[i][0] = b0; bx[i][1] = b1; bx[i][2] = b2; bx[i][3] = b3;
        ar[i] = (b2 - b0) * (b3 - b1);
        float s = scoresT[cls * NUM_ROIS + i];
        sc[i] = (s > 0.05f) ? s : -1.0f;
    }
    __syncthreads();
    for (int it = 0; it < MAX_DET; it++) {
        float bv = -1e30f; int bi = 0;
        for (int i = tid; i < NUM_ROIS; i += 256) {
            float v = sc[i];
            if (v > bv || (v == bv && i < bi)) { bv = v; bi = i; }
        }
        rv[tid] = bv; ri[tid] = bi; __syncthreads();
        for (int s2 = 128; s2 > 0; s2 >>= 1) {
            if (tid < s2) {
                if (rv[tid + s2] > rv[tid] ||
                    (rv[tid + s2] == rv[tid] && ri[tid + s2] < ri[tid])) {
                    rv[tid] = rv[tid + s2]; ri[tid] = ri[tid + s2];
                }
            }
            __syncthreads();
        }
        int best = ri[0];
        if (tid == 0) {
            selI[cls * MAX_DET + it] = best;
            selS[cls * MAX_DET + it] = rv[0];
        }
        float b0 = bx[best][0], b1 = bx[best][1], b2 = bx[best][2], b3 = bx[best][3];
        float ab = ar[best];
        for (int i = tid; i < NUM_ROIS; i += 256) {
            float xx1 = fmaxf(b0, bx[i][0]), yy1 = fmaxf(b1, bx[i][1]);
            float xx2 = fminf(b2, bx[i][2]), yy2 = fminf(b3, bx[i][3]);
            float inter = fmaxf(xx2 - xx1, 0.0f) * fmaxf(yy2 - yy1, 0.0f);
            float iou = inter / (ab + ar[i] - inter + 1e-6f);
            if (iou > 0.5f || i == best) sc[i] = -1.0f;
        }
        __syncthreads();
    }
}

// ---------------------------------------------------------------------------
// Global top-100 over 8000 candidates + output packing
// out: [0]=num_det, [1..400]=boxes, [401..500]=scores, [501..600]=classes
// ---------------------------------------------------------------------------
__global__ __launch_bounds__(256)
void topk_kernel(const float* __restrict__ selS, const int* __restrict__ selI,
                 const float* __restrict__ bboxes, float* __restrict__ out) {
    __shared__ float s[NUM_CLS * MAX_DET];
    __shared__ float rv[256];
    __shared__ int   ri[256];
    __shared__ int   cntSh;
    int tid = threadIdx.x;
    for (int i = tid; i < NUM_CLS * MAX_DET; i += 256) s[i] = selS[i];
    if (tid == 0) cntSh = 0;
    __syncthreads();
    for (int it = 0; it < MAX_DET; it++) {
        float bv = -1e30f; int bi = 0;
        for (int i = tid; i < NUM_CLS * MAX_DET; i += 256) {
            float v = s[i];
            if (v > bv || (v == bv && i < bi)) { bv = v; bi = i; }
        }
        rv[tid] = bv; ri[tid] = bi; __syncthreads();
        for (int st = 128; st > 0; st >>= 1) {
            if (tid < st) {
                if (rv[tid + st] > rv[tid] ||
                    (rv[tid + st] == rv[tid] && ri[tid + st] < ri[tid])) {
                    rv[tid] = rv[tid + st]; ri[tid] = ri[tid + st];
                }
            }
            __syncthreads();
        }
        if (tid == 0) {
            int g = ri[0]; float val = rv[0];
            int valid = (val > 0.0f) ? 1 : 0;
            cntSh += valid;
            int cls = g / MAX_DET;
            int bidx = selI[g];
            out[1 + it * 4 + 0] = valid ? bboxes[bidx * 4 + 0] : 0.0f;
            out[1 + it * 4 + 1] = valid ? bboxes[bidx * 4 + 1] : 0.0f;
            out[1 + it * 4 + 2] = valid ? bboxes[bidx * 4 + 2] : 0.0f;
            out[1 + it * 4 + 3] = valid ? bboxes[bidx * 4 + 3] : 0.0f;
            out[401 + it] = valid ? val : 0.0f;
            out[501 + it] = valid ? (float)cls : -1.0f;
            s[g] = -1e30f;
        }
        __syncthreads();
    }
    if (tid == 0) out[0] = (float)cntSh;
}

// ---------------------------------------------------------------------------
// Launcher
// ---------------------------------------------------------------------------
extern "C" void kernel_launch(void* const* d_in, const int* in_sizes, int n_in,
                              void* d_out, int out_size, void* d_ws, size_t ws_size,
                              hipStream_t stream) {
    (void)in_sizes; (void)n_in; (void)out_size; (void)ws_size;
    const float* feat0 = (const float*)d_in[0];
    const float* feat1 = (const float*)d_in[1];
    const float* feat2 = (const float*)d_in[2];
    const float* feat3 = (const float*)d_in[3];
    const float* proposals = (const float*)d_in[4];
    const float* fc1_w = (const float*)d_in[5];
    const float* fc1_b = (const float*)d_in[6];
    const float* fc2_w = (const float*)d_in[7];
    const float* fc2_b = (const float*)d_in[8];
    const float* cls_w = (const float*)d_in[9];
    const float* cls_b = (const float*)d_in[10];
    const float* reg_w = (const float*)d_in[11];
    const float* reg_b = (const float*)d_in[12];
    float* out = (float*)d_out;

    char* ws = (char*)d_ws;
    size_t off = 0;
    auto alloc = [&](size_t bytes) -> void* {
        void* p = (void*)(ws + off);
        off = (off + bytes + 255) & ~(size_t)255;
        return p;
    };
    unsigned short* X   = (unsigned short*)alloc((size_t)NUM_ROIS * FEAT_D * 2);
    unsigned short* H1  = (unsigned short*)alloc((size_t)NUM_ROIS * FC_DIM * 2);
    unsigned short* H2  = (unsigned short*)alloc((size_t)NUM_ROIS * FC_DIM * 2);
    float* clsSum       = (float*)alloc((size_t)NUM_ROIS * NUM_CLS1 * 4);
    float* bboxPred     = (float*)alloc((size_t)NUM_ROIS * 4 * 4);
    float* roisA        = (float*)alloc((size_t)NUM_ROIS * 4 * 4);
    float* roisB        = (float*)alloc((size_t)NUM_ROIS * 4 * 4);
    float* bboxes       = (float*)alloc((size_t)NUM_ROIS * 4 * 4);
    float* scoresT      = (float*)alloc((size_t)NUM_CLS * NUM_ROIS * 4);
    int*   selI         = (int*)alloc((size_t)NUM_CLS * MAX_DET * 4);
    float* selS         = (float*)alloc((size_t)NUM_CLS * MAX_DET * 4);

    const float STDS[3][4] = {{0.10f, 0.10f, 0.20f, 0.20f},
                              {0.05f, 0.05f, 0.10f, 0.10f},
                              {0.033f, 0.033f, 0.067f, 0.067f}};

    const float* roisCur = proposals;
    float* roisBuf[2] = {roisA, roisB};

    for (int s = 0; s < 3; s++) {
        roi_align_kernel<<<NUM_ROIS, 256, 0, stream>>>(feat0, feat1, feat2, feat3,
                                                       roisCur, X);
        gemm_bf16_kernel<<<dim3(FC_DIM / 64, NUM_ROIS / 64), 128, 0, stream>>>(
            X, fc1_w + (size_t)s * FEAT_D * FC_DIM, fc1_b + (size_t)s * FC_DIM,
            H1, NUM_ROIS, FEAT_D, FC_DIM, 1);
        gemm_bf16_kernel<<<dim3(FC_DIM / 64, NUM_ROIS / 64), 128, 0, stream>>>(
            H1, fc2_w + (size_t)s * FC_DIM * FC_DIM, fc2_b + (size_t)s * FC_DIM,
            H2, NUM_ROIS, FC_DIM, FC_DIM, 1);
        heads_kernel<<<NUM_ROIS, 256, 0, stream>>>(
            H2, cls_w + (size_t)s * FC_DIM * NUM_CLS1, cls_b + (size_t)s * NUM_CLS1,
            reg_w + (size_t)s * FC_DIM * 4, reg_b + (size_t)s * 4,
            clsSum, bboxPred, s);
        if (s < 2) {
            delta2bbox_kernel<<<(NUM_ROIS + 255) / 256, 256, 0, stream>>>(
                roisCur, bboxPred, roisBuf[s],
                STDS[s][0], STDS[s][1], STDS[s][2], STDS[s][3]);
            roisCur = roisBuf[s];
        }
    }

    softmax_final_kernel<<<(NUM_ROIS + 127) / 128, 128, 0, stream>>>(
        clsSum, roisCur, bboxPred, scoresT, bboxes);
    nms_kernel<<<NUM_CLS, 256, 0, stream>>>(bboxes, scoresT, selI, selS);
    topk_kernel<<<1, 256, 0, stream>>>(selS, selI, bboxes, out);
}
